// RPN_11476152615331
// MI455X (gfx1250) — matrix operand order repaired and checked by reading, not compiled, not run on hardware
//
#include <hip/hip_runtime.h>
#include <stdint.h>

typedef __attribute__((ext_vector_type(16))) _Float16 v16h;
typedef __attribute__((ext_vector_type(8)))  _Float16 v8h;
typedef __attribute__((ext_vector_type(8)))  float    v8f;

#define S_DIM   160
#define P_TOT   (S_DIM * S_DIM)      // 25600 spatial positions
#define C_IN    1024
#define OUT_DIM 24
#define KTOT    (9 * C_IN)           // 9216
#define KSTEPS  (KTOT / 32)          // 288
#define MTILES  (P_TOT / 16)         // 1600
#define HPAD    32                   // hidden channels padded to one WMMA K-step
#define NCH     64                   // 18 cls + 36 reg = 54 -> padded 64
#define PRE_NMS 2000
#define NMS_THR 0.7f
#define MIN_SZ  16.0f
#define IMG_M1  2559.0f
#define MAX_ANCH 230400ull           // S*S*9

// B staging geometry
#define FRAG_H   24                              // padded fragment halves (48B stride)
#define CHUNK_KS 8                               // k-steps per LDS chunk
#define CHUNK_HALVES (CHUNK_KS * 2 * 32 * FRAG_H) // 12288 halves = 24 KB
#define NCHUNKS  (KSTEPS / CHUNK_KS)             // 36

#if defined(__has_builtin)
#if __has_builtin(__builtin_amdgcn_global_load_async_to_lds_b128)
#define HAVE_ASYNC_LDS 1
#endif
#endif

#ifdef HAVE_ASYNC_LDS
// Exact pointee type from the hipcc diagnostic: int __attribute__((vector_size(16)))
typedef int v4i_t __attribute__((vector_size(16)));
typedef __attribute__((address_space(1))) v4i_t* gv4p;   // global (__device__) int4*
typedef __attribute__((address_space(3))) v4i_t* lv4p;   // LDS int4*
# if __has_builtin(__builtin_amdgcn_s_wait_asynccnt)
#  define WAIT_ASYNC(n) __builtin_amdgcn_s_wait_asynccnt(n)
# else
#  define WAIT_ASYNC(n) asm volatile("s_wait_asynccnt %0" :: "i"(n) : "memory")
# endif
#else
# define WAIT_ASYNC(n) ((void)0)
#endif

// ---------------- workspace layout (bytes) ----------------
static const size_t OFF_XT     = 0;
static const size_t SZ_XT      = (size_t)P_TOT * C_IN * 2;            // 52,428,800
static const size_t OFF_BPACK  = OFF_XT + SZ_XT;
static const size_t SZ_BPACK   = (size_t)KSTEPS * 2 * 32 * FRAG_H * 2; // 884,736
static const size_t OFF_BPACK2 = OFF_BPACK + SZ_BPACK;
static const size_t SZ_BPACK2  = 4096;
static const size_t OFF_HT     = OFF_BPACK2 + SZ_BPACK2;
static const size_t SZ_HT      = (size_t)P_TOT * HPAD * 2;            // 1,638,400
static const size_t OFF_CLSREG = OFF_HT + SZ_HT;
static const size_t SZ_CLSREG  = (size_t)P_TOT * NCH * 4;             // 6,553,600
static const size_t OFF_ZERO   = OFF_CLSREG + SZ_CLSREG;              // 2 KB zero page
// Overlay the xT region; only written AFTER the conv GEMM consumed xT.
static const size_t OFF_BBOX   = 0;
static const size_t OFF_SCORE  = MAX_ANCH * 16;
static const size_t OFF_KEYS   = OFF_SCORE + MAX_ANCH * 4;

// ---------------- tiny utility kernels ----------------
__global__ void k_zero16(_Float16* z) {
  for (int i = threadIdx.x; i < 1024; i += 256) z[i] = (_Float16)0.f;
}

// x (C_IN, P_TOT) f32  ->  xT (P_TOT, C_IN) f16, LDS-tiled transpose
__global__ void k_xpose(const float* __restrict__ x, _Float16* __restrict__ xT) {
  __shared__ _Float16 tile[32][33];
  int p0 = blockIdx.x * 32, c0 = blockIdx.y * 32;
  int tx = threadIdx.x, ty = threadIdx.y;
#pragma unroll
  for (int cc = 0; cc < 4; ++cc) {
    int cl = ty + cc * 8;
    tile[cl][tx] = (_Float16)x[(size_t)(c0 + cl) * P_TOT + p0 + tx];
  }
  __syncthreads();
#pragma unroll
  for (int pp = 0; pp < 4; ++pp) {
    int pl = ty + pp * 8;
    xT[(size_t)(p0 + pl) * C_IN + c0 + tx] = tile[tx][pl];
  }
}

// conv_w (24,1024,3,3) f32 -> B-fragments f16 [ks][nt][lane][FRAG_H]
// element e<16 of a lane's fragment is Klocal = (lane>>4)*16 + e; e>=16 is pad.
__global__ void k_packw(const float* __restrict__ w, _Float16* __restrict__ bp) {
  int tid = blockIdx.x * blockDim.x + threadIdx.x;   // 288*2*32*24 = 442368
  int e = tid % FRAG_H;
  int r = tid / FRAG_H;
  int lane = r & 31; r >>= 5;
  int nt = r & 1;
  int ks = r >> 1;
  float v = 0.f;
  if (e < 16) {
    int o = nt * 16 + (lane & 15);
    int K = ks * 32 + (lane >> 4) * 16 + e;          // K = tap*1024 + ci
    int tap = K >> 10;
    int ci  = K & 1023;
    if (o < OUT_DIM) v = w[((size_t)o * C_IN + ci) * 9 + tap];
  }
  bp[tid] = (_Float16)v;
}

// cls_w (18,24) + reg_w (36,24) -> B-fragments f16 [nt(4)][lane][16], K=o padded 32
__global__ void k_packhead(const float* __restrict__ cw, const float* __restrict__ rw,
                           _Float16* __restrict__ bp) {
  int tid = blockIdx.x * blockDim.x + threadIdx.x;   // 2048 threads
  int e = tid & 15;
  int lane = (tid >> 4) & 31;
  int nt = tid >> 9;
  int ch = nt * 16 + (lane & 15);
  int o = (lane >> 4) * 16 + e;
  float v = 0.f;
  if (o < OUT_DIM) {
    if (ch < 18)      v = cw[ch * OUT_DIM + o];
    else if (ch < 54) v = rw[(ch - 18) * OUT_DIM + o];
  }
  bp[tid] = (_Float16)v;
}

// Cooperative 24KB chunk stage: 96 B per thread, 6x b128 async copies.
__device__ __forceinline__ void stage_chunk(const _Float16* __restrict__ gsrc,
                                            _Float16* lbuf, int tid) {
#ifdef HAVE_ASYNC_LDS
  gv4p g = (gv4p)(gsrc + (size_t)tid * 48);   // 48 halves = 96 B = 6 x int4
  lv4p l = (lv4p)(lbuf + (size_t)tid * 48);
#pragma unroll
  for (int i = 0; i < 6; ++i)
    __builtin_amdgcn_global_load_async_to_lds_b128(g + i, l + i, 0, 0);
#else
  const v8h* s = (const v8h*)(gsrc) + tid * 6;
  v8h* d = (v8h*)(lbuf) + tid * 6;
#pragma unroll
  for (int i = 0; i < 6; ++i) d[i] = s[i];
#endif
}

// ---------------- 3x3 conv as implicit GEMM via WMMA + async B staging ----------------
__global__ void __launch_bounds__(256) k_conv(const _Float16* __restrict__ xT,
                                              const _Float16* __restrict__ bp,
                                              const float* __restrict__ bias,
                                              const _Float16* __restrict__ zbuf,
                                              _Float16* __restrict__ hT) {
  __shared__ __align__(16) _Float16 sB[2][CHUNK_HALVES];
  int tid = threadIdx.x;
  int lane = tid & 31;
  int wave = tid >> 5;
  int mtile = blockIdx.x * 8 + wave;
  int mrow = lane & 15, half = lane >> 4;
  int p = mtile * 16 + mrow;
  int py = p / S_DIM, px = p % S_DIM;

  v8f acc0 = {0.f,0.f,0.f,0.f,0.f,0.f,0.f,0.f};
  v8f acc1 = {0.f,0.f,0.f,0.f,0.f,0.f,0.f,0.f};

  stage_chunk(bp, sB[0], tid);                       // chunk 0 in flight

  for (int c = 0; c < NCHUNKS; ++c) {
    if (c + 1 < NCHUNKS) {
      stage_chunk(bp + (size_t)(c + 1) * CHUNK_HALVES, sB[(c + 1) & 1], tid);
      WAIT_ASYNC(6);                                 // drain chunk c, keep c+1 in flight
    } else {
      WAIT_ASYNC(0);
    }
    __syncthreads();

    int tap = c >> 2;
    int kcbase = (c & 3) * CHUNK_KS;
    int yy = py + tap / 3 - 1;
    int xx = px + tap % 3 - 1;
    bool ok = ((unsigned)yy < (unsigned)S_DIM) && ((unsigned)xx < (unsigned)S_DIM);
    const _Float16* aptr = ok ? (xT + (size_t)(yy * S_DIM + xx) * C_IN) : zbuf;
    __builtin_prefetch((const void*)aptr, 0, 1);     // global_prefetch_b8
    const _Float16* lb = sB[c & 1];

#pragma unroll
    for (int kcl = 0; kcl < CHUNK_KS; ++kcl) {
      int ci0 = (kcbase + kcl) * 32 + half * 16;
      union { v16h v; v8h h[2]; } a, f0, f1;
      a.h[0] = *(const v8h*)(aptr + ci0);
      a.h[1] = *(const v8h*)(aptr + ci0 + 8);
      const _Float16* fb = lb + ((size_t)(kcl * 2) * 32 + lane) * FRAG_H;
      f0.h[0] = *(const v8h*)(fb);
      f0.h[1] = *(const v8h*)(fb + 8);
      f1.h[0] = *(const v8h*)(fb + 32 * FRAG_H);
      f1.h[1] = *(const v8h*)(fb + 32 * FRAG_H + 8);
      acc0 = __builtin_amdgcn_wmma_f32_16x16x32_f16(false, a.v, false, f0.v,
                                                    (short)0, acc0, false, false);
      acc1 = __builtin_amdgcn_wmma_f32_16x16x32_f16(false, a.v, false, f1.v,
                                                    (short)0, acc1, false, false);
    }
    __syncthreads();
  }

  // D layout: VGPR r -> M = r + 8*half, N = lane&15
  int n = lane & 15;
  float bn0 = bias[n];
  float bn1 = (n + 16 < OUT_DIM) ? bias[n + 16] : 0.f;
#pragma unroll
  for (int r = 0; r < 8; ++r) {
    int hp = mtile * 16 + r + 8 * half;
    float v0 = acc0[r] + bn0;  v0 = v0 > 0.f ? v0 : 0.f;
    hT[(size_t)hp * HPAD + n] = (_Float16)v0;
    float v1 = 0.f;
    if (n + 16 < OUT_DIM) { v1 = acc1[r] + bn1; v1 = v1 > 0.f ? v1 : 0.f; }
    hT[(size_t)hp * HPAD + n + 16] = (_Float16)v1;
  }
}

// ---------------- 1x1 cls/reg heads: K=32, one WMMA per N-tile ----------------
__global__ void __launch_bounds__(256) k_head(const _Float16* __restrict__ hT,
                                              const _Float16* __restrict__ bp,
                                              const float* __restrict__ cb,
                                              const float* __restrict__ rb,
                                              float* __restrict__ clsreg) {
  int lane = threadIdx.x & 31;
  int wave = threadIdx.x >> 5;
  int mtile = blockIdx.x * 8 + wave;
  int n = lane & 15, half = lane >> 4;
  int p = mtile * 16 + n;   // A-row for this lane

  union { v16h v; v8h h[2]; } a;
  a.h[0] = *(const v8h*)(hT + (size_t)p * HPAD + half * 16);
  a.h[1] = *(const v8h*)(hT + (size_t)p * HPAD + half * 16 + 8);

  v8f acc[4];
#pragma unroll
  for (int t = 0; t < 4; ++t) {
    union { v16h v; v8h h[2]; } b;
    b.h[0] = *(const v8h*)(bp + ((size_t)t * 32 + lane) * 16);
    b.h[1] = *(const v8h*)(bp + ((size_t)t * 32 + lane) * 16 + 8);
    v8f c = {0.f,0.f,0.f,0.f,0.f,0.f,0.f,0.f};
    acc[t] = __builtin_amdgcn_wmma_f32_16x16x32_f16(false, a.v, false, b.v,
                                                    (short)0, c, false, false);
  }
#pragma unroll
  for (int t = 0; t < 4; ++t) {
    int ch = t * 16 + n;
    float bv = 0.f;
    if (ch < 18)      bv = cb[ch];
    else if (ch < 54) bv = rb[ch - 18];
#pragma unroll
    for (int r = 0; r < 8; ++r) {
      int hp = mtile * 16 + r + 8 * half;
      clsreg[(size_t)hp * NCH + ch] = acc[t][r] + bv;
    }
  }
}

// ---------------- anchor decode, scoring, sort keys ----------------
__global__ void k_decode(const float* __restrict__ clsreg,
                         const float* __restrict__ anchors,
                         const int* __restrict__ vidx, int nvalid,
                         float* __restrict__ out_prop, float* __restrict__ out_cls,
                         float* __restrict__ bboxes, float* __restrict__ scores,
                         unsigned long long* __restrict__ keys) {
  int i = blockIdx.x * blockDim.x + threadIdx.x;
  if (i >= nvalid) return;
  int j = vidx[i];
  int p = j / 9, an = j - p * 9;
  const float* cr = clsreg + (size_t)p * NCH;
  float c0 = cr[2 * an], c1 = cr[2 * an + 1];
  float t0 = cr[18 + 4 * an + 0], t1 = cr[18 + 4 * an + 1];
  float t2 = cr[18 + 4 * an + 2], t3 = cr[18 + 4 * an + 3];
  float acx = anchors[(size_t)j * 4 + 0], acy = anchors[(size_t)j * 4 + 1];
  float aw  = anchors[(size_t)j * 4 + 2], ah  = anchors[(size_t)j * 4 + 3];
  float pxc = acx + aw * t0, pyc = acy + ah * t1;
  float pw = aw * expf(t2), ph = ah * expf(t3);
  out_prop[(size_t)i * 4 + 0] = pxc;  out_prop[(size_t)i * 4 + 1] = pyc;
  out_prop[(size_t)i * 4 + 2] = pw;   out_prop[(size_t)i * 4 + 3] = ph;
  out_cls[(size_t)i * 2 + 0] = c0;    out_cls[(size_t)i * 2 + 1] = c1;
  float rx0 = pxc - 0.5f * (pw - 1.f), ry0 = pyc - 0.5f * (ph - 1.f);
  float rx1 = rx0 + pw - 1.f,          ry1 = ry0 + ph - 1.f;
  float x0 = fminf(fmaxf(rx0, 0.f), IMG_M1), y0 = fminf(fmaxf(ry0, 0.f), IMG_M1);
  float x1 = fminf(fmaxf(rx1, 0.f), IMG_M1), y1 = fminf(fmaxf(ry1, 0.f), IMG_M1);
  float prob = 1.f / (1.f + expf(c0 - c1));                  // softmax class-1
  bool fm = (x1 - x0 + 1.f >= MIN_SZ) && (y1 - y0 + 1.f >= MIN_SZ);
  float sc = fm ? prob : -1.f;
  bboxes[(size_t)i * 4 + 0] = x0; bboxes[(size_t)i * 4 + 1] = y0;
  bboxes[(size_t)i * 4 + 2] = x1; bboxes[(size_t)i * 4 + 3] = y1;
  scores[i] = sc;
  unsigned b = __float_as_uint(sc);
  b = (b & 0x80000000u) ? ~b : (b | 0x80000000u);            // order-preserving
  keys[i] = ((unsigned long long)b << 32) | (unsigned)(~(unsigned)i);
}

__global__ void k_fillpad(unsigned long long* keys, int nvalid, int npow2) {
  int i = blockIdx.x * blockDim.x + threadIdx.x + nvalid;
  if (i < npow2) keys[i] = 0ull;
}

__global__ void k_bitonic(unsigned long long* __restrict__ keys,
                          unsigned jj, unsigned kk) {
  unsigned i = blockIdx.x * blockDim.x + threadIdx.x;
  unsigned ixj = i ^ jj;
  if (ixj > i) {
    unsigned long long a = keys[i], b = keys[ixj];
    bool up = ((i & kk) == 0);
    bool sw = up ? (a < b) : (a > b);                        // descending sort
    if (sw) { keys[i] = b; keys[ixj] = a; }
  }
}

// ---------------- greedy NMS, single workgroup, all state in LDS ----------------
__global__ void __launch_bounds__(256) k_nms(const unsigned long long* __restrict__ keys,
                                             const float* __restrict__ scores,
                                             const float* __restrict__ bboxes,
                                             int nvalid,
                                             float* __restrict__ out_filt,
                                             float* __restrict__ out_prob) {
  __shared__ float sx0[PRE_NMS], sy0[PRE_NMS], sx1[PRE_NMS], sy1[PRE_NMS];
  __shared__ float sAr[PRE_NMS], sV[PRE_NMS];
  __shared__ int   sKeep[PRE_NMS];
  int tid = threadIdx.x;
  for (int t = tid; t < PRE_NMS; t += 256) {
    unsigned long long key = keys[t];
    unsigned idx = ~((unsigned)(key & 0xFFFFFFFFull));
    float v = 0.f, x0 = 0.f, y0 = 0.f, x1 = 0.f, y1 = 0.f;
    if ((key >> 32) != 0ull && idx < (unsigned)nvalid) {
      v = scores[idx];
      x0 = bboxes[(size_t)idx * 4 + 0]; y0 = bboxes[(size_t)idx * 4 + 1];
      x1 = bboxes[(size_t)idx * 4 + 2]; y1 = bboxes[(size_t)idx * 4 + 3];
    }
    sx0[t] = x0; sy0[t] = y0; sx1[t] = x1; sy1[t] = y1;
    sAr[t] = (x1 - x0 + 1.f) * (y1 - y0 + 1.f);
    sV[t] = v;
    sKeep[t] = (v > 0.f) ? 1 : 0;
  }
  __syncthreads();
  for (int i = 0; i < PRE_NMS; ++i) {
    if (sKeep[i]) {
      float bx0 = sx0[i], by0 = sy0[i], bx1 = sx1[i], by1 = sy1[i], ba = sAr[i];
      for (int j = i + 1 + tid; j < PRE_NMS; j += 256) {
        float xx0 = fmaxf(bx0, sx0[j]), yy0 = fmaxf(by0, sy0[j]);
        float xx1 = fminf(bx1, sx1[j]), yy1 = fminf(by1, sy1[j]);
        float inter = fmaxf(0.f, xx1 - xx0 + 1.f) * fmaxf(0.f, yy1 - yy0 + 1.f);
        float iou = inter / (ba + sAr[j] - inter);
        if (iou > NMS_THR) sKeep[j] = 0;
      }
    }
    __syncthreads();
  }
  for (int t = tid; t < PRE_NMS; t += 256) {
    float k = sKeep[t] ? 1.f : 0.f;
    float bw = sx1[t] - sx0[t] + 1.f, bh = sy1[t] - sy0[t] + 1.f;
    float bcx = sx0[t] + 0.5f * (bw - 1.f), bcy = sy0[t] + 0.5f * (bh - 1.f);
    out_filt[(size_t)t * 4 + 0] = bcx * k;
    out_filt[(size_t)t * 4 + 1] = bcy * k;
    out_filt[(size_t)t * 4 + 2] = bw * k;
    out_filt[(size_t)t * 4 + 3] = bh * k;
    out_prob[t] = sKeep[t] ? sV[t] : 0.f;
  }
}

// ---------------- host side ----------------
extern "C" void kernel_launch(void* const* d_in, const int* in_sizes, int n_in,
                              void* d_out, int out_size, void* d_ws, size_t ws_size,
                              hipStream_t stream) {
  (void)n_in; (void)out_size; (void)ws_size;
  const float* x      = (const float*)d_in[0];
  const float* conv_w = (const float*)d_in[1];
  const float* conv_b = (const float*)d_in[2];
  const float* cls_w  = (const float*)d_in[3];
  const float* cls_b  = (const float*)d_in[4];
  const float* reg_w  = (const float*)d_in[5];
  const float* reg_b  = (const float*)d_in[6];
  const float* anchors = (const float*)d_in[7];
  const int*   vidx   = (const int*)d_in[8];
  int nvalid = in_sizes[8];

  float* out = (float*)d_out;
  char* ws = (char*)d_ws;
  _Float16* xT     = (_Float16*)(ws + OFF_XT);
  _Float16* bpack  = (_Float16*)(ws + OFF_BPACK);
  _Float16* bpack2 = (_Float16*)(ws + OFF_BPACK2);
  _Float16* hT     = (_Float16*)(ws + OFF_HT);
  float*    clsreg = (float*)   (ws + OFF_CLSREG);
  _Float16* zbuf   = (_Float16*)(ws + OFF_ZERO);
  float*    bboxes = (float*)   (ws + OFF_BBOX);
  float*    scores = (float*)   (ws + OFF_SCORE);
  unsigned long long* keys = (unsigned long long*)(ws + OFF_KEYS);

  k_zero16<<<1, 256, 0, stream>>>(zbuf);
  k_xpose<<<dim3(P_TOT / 32, C_IN / 32), dim3(32, 8), 0, stream>>>(x, xT);
  k_packw<<<(KSTEPS * 2 * 32 * FRAG_H) / 256, 256, 0, stream>>>(conv_w, bpack);
  k_packhead<<<2048 / 256, 256, 0, stream>>>(cls_w, reg_w, bpack2);
  k_conv<<<MTILES / 8, 256, 0, stream>>>(xT, bpack, conv_b, zbuf, hT);
  k_head<<<MTILES / 8, 256, 0, stream>>>(hT, bpack2, cls_b, reg_b, clsreg);
  k_decode<<<(nvalid + 255) / 256, 256, 0, stream>>>(
      clsreg, anchors, vidx, nvalid,
      out, out + (size_t)nvalid * 4, bboxes, scores, keys);

  unsigned npow2 = 2048;
  while (npow2 < (unsigned)nvalid) npow2 <<= 1;
  int padN = (int)npow2 - nvalid;
  if (padN > 0)
    k_fillpad<<<(padN + 255) / 256, 256, 0, stream>>>(keys, nvalid, (int)npow2);
  for (unsigned k = 2; k <= npow2; k <<= 1)
    for (unsigned j = k >> 1; j >= 1; j >>= 1)
      k_bitonic<<<npow2 / 256, 256, 0, stream>>>(keys, j, k);

  k_nms<<<1, 256, 0, stream>>>(keys, scores, bboxes, nvalid,
                               out + (size_t)nvalid * 6,
                               out + (size_t)nvalid * 6 + PRE_NMS * 4);
}
